// VarianceAdaptor_70068096468014
// MI455X (gfx1250) — matrix operand order repaired and checked by reading, not compiled, not run on hardware
//
#include <hip/hip_runtime.h>
#include <hip/hip_bf16.h>
#include <math.h>

typedef _Float16 half_t;
typedef __attribute__((ext_vector_type(16))) _Float16 v16h;
typedef __attribute__((ext_vector_type(8)))  _Float16 v8h;
typedef __attribute__((ext_vector_type(8)))  float    v8f;

#define B_SZ   32
#define T_IN   256
#define D_SZ   256
#define MEL    2048
#define NBINS  256

// ---------------------------------------------------------------------------
// Fused conv1d(K=3, 256->256, pad=1) + bias + LayerNorm + ReLU.
// f16 in, f16 out (LN math in f32). Block = 128 threads (4 waves); one
// 16(time) x 256(cout) tile per block. GEMM over K=768 in 24 chunks of 32
// via v_wmma_f32_16x16x32_f16; each wave owns 4 n-tiles.
// ---------------------------------------------------------------------------
__global__ __launch_bounds__(128)
void conv_ln_relu_kernel(const half_t* __restrict__ in,   // [B,T,256] f16
                         const float* __restrict__ w,     // [3,256,256] f32
                         const float* __restrict__ bias,  // [256]
                         const float* __restrict__ gamma, // [256]
                         const float* __restrict__ beta,  // [256]
                         half_t* __restrict__ out,        // [B,T,256] f16
                         int T)
{
    __shared__ __align__(16) half_t xs[18 * 256];   // input tile rows t0-1..t0+16
    __shared__ __align__(16) half_t wsT[256 * 32];  // weight chunk [cout][kLocal]
    __shared__ float  hs[16 * 256];                 // conv output tile f32
    __shared__ float  mu[16], rs[16];

    const int tid     = threadIdx.x;
    const int lane    = tid & 31;
    const int wid     = tid >> 5;
    const int halfSel = lane >> 4;
    const int l15     = lane & 15;
    const int t0      = blockIdx.x * 16;
    const int b       = blockIdx.y;

    // ---- stage haloed input tile: pure 16B copies ----------------------
    for (int i = tid; i < 18 * 32; i += 128) {
        int r = i >> 5, c8 = (i & 31) << 3;
        int t = t0 + r - 1;
        v8h v;
#pragma unroll
        for (int e = 0; e < 8; ++e) v[e] = (half_t)0;
        if (t >= 0 && t < T)
            v = *(const v8h*)&in[((size_t)b * T + t) * D_SZ + c8];
        *(v8h*)&xs[r * 256 + c8] = v;
    }

    v8f acc[4];
#pragma unroll
    for (int nt = 0; nt < 4; ++nt)
#pragma unroll
        for (int j = 0; j < 8; ++j) acc[nt][j] = 0.0f;

    // ---- GEMM main loop: 24 chunks of K=32 -----------------------------
    for (int kk = 0; kk < 24; ++kk) {
        const int tap  = kk >> 3;
        const int cin0 = (kk & 7) << 5;

        __syncthreads();  // previous chunk's wsT reads complete (covers xs too)
        for (int i = tid; i < 32 * 256; i += 128) {
            int kl = i >> 8, co = i & 255;
            wsT[co * 32 + kl] =
                (half_t)w[((size_t)(tap * D_SZ) + cin0 + kl) * D_SZ + co];
        }
        __syncthreads();

        // prefetch next chunk's weights into cache (global_prefetch_b8)
        if (kk < 23) {
            const int ntap  = (kk + 1) >> 3;
            const int ncin0 = ((kk + 1) & 7) << 5;
            const float* nw = &w[((size_t)(ntap * D_SZ) + ncin0) * D_SZ];
            __builtin_prefetch(nw + (size_t)tid * 64, 0, 1);
        }

        // A fragment: M = l15, K(e) = (e/8)*16 + halfSel*8 + e%8
        v16h afrag;
        {
            const half_t* xrow = &xs[(l15 + tap) * 256 + cin0 + halfSel * 8];
#pragma unroll
            for (int e = 0; e < 8; ++e) afrag[e] = xrow[e];
#pragma unroll
            for (int e = 0; e < 8; ++e) afrag[8 + e] = xrow[16 + e];
        }

#pragma unroll
        for (int nt = 0; nt < 4; ++nt) {
            const int n0 = ((wid << 2) + nt) << 4;
            // B fragment: N = n0 + l15, K(e) = (e/8)*16 + halfSel*8 + e%8
            v16h bfrag;
            const half_t* wrow = &wsT[(n0 + l15) * 32 + halfSel * 8];
#pragma unroll
            for (int e = 0; e < 8; ++e) bfrag[e] = wrow[e];
#pragma unroll
            for (int e = 0; e < 8; ++e) bfrag[8 + e] = wrow[16 + e];

            acc[nt] = __builtin_amdgcn_wmma_f32_16x16x32_f16(
                false, afrag, false, bfrag, (short)0, acc[nt], false, false);
        }
    }

    // ---- epilogue: bias, scatter D to LDS, LayerNorm, ReLU -------------
    __syncthreads();
#pragma unroll
    for (int nt = 0; nt < 4; ++nt) {
        const int n  = (((wid << 2) + nt) << 4) + l15;
        const float bv = bias[n];
#pragma unroll
        for (int v = 0; v < 8; ++v) {
            int m = v + halfSel * 8;               // D layout: M = v + (lane/16)*8
            hs[m * 256 + n] = acc[nt][v] + bv;
        }
    }
    __syncthreads();

    if (tid < 16) {
        float s = 0.0f, s2 = 0.0f;
        for (int f = 0; f < 256; ++f) {
            float v = hs[tid * 256 + f];
            s += v; s2 += v * v;
        }
        float m   = s * (1.0f / 256.0f);
        float var = s2 * (1.0f / 256.0f) - m * m;
        mu[tid] = m;
        rs[tid] = rsqrtf(var + 1e-5f);
    }
    __syncthreads();

    for (int i = tid; i < 16 * 256; i += 128) {
        int m = i >> 8, f = i & 255;
        float v = (hs[i] - mu[m]) * rs[m] * gamma[f] + beta[f];
        out[((size_t)b * T + t0 + m) * D_SZ + f] = (half_t)fmaxf(v, 0.0f);
    }
}

// ---------------------------------------------------------------------------
// row(f16) . wl + bl -> relu [-> exp()-1].  One wave per 256-wide row.
// ---------------------------------------------------------------------------
__global__ __launch_bounds__(256)
void rowdot_kernel(const half_t* __restrict__ in, const float* __restrict__ wl,
                   const float* __restrict__ bl, float* __restrict__ out,
                   int nrows, int expMode)
{
    const int lane = threadIdx.x & 31;
    const int row  = blockIdx.x * (blockDim.x >> 5) + (threadIdx.x >> 5);
    if (row >= nrows) return;
    const half_t* r = in + (size_t)row * 256;
    float s = 0.0f;
#pragma unroll
    for (int i = 0; i < 8; ++i) s += (float)r[lane * 8 + i] * wl[lane * 8 + i];
#pragma unroll
    for (int off = 16; off >= 1; off >>= 1) s += __shfl_xor(s, off, 32);
    if (lane == 0) {
        float v = fmaxf(s + bl[0], 0.0f);
        out[row] = expMode ? (expf(v) - 1.0f) : v;
    }
}

// ---------------------------------------------------------------------------
__global__ void f32_to_f16_kernel(const float* __restrict__ in,
                                  half_t* __restrict__ out, int n)
{
    int i = blockIdx.x * blockDim.x + threadIdx.x;
    if (i < n) out[i] = (half_t)in[i];
}

// ---------------------------------------------------------------------------
__global__ void cumsum_kernel(const int* __restrict__ dur, int* __restrict__ cum)
{
    if (threadIdx.x == 0) {
        const int b = blockIdx.x;
        int s = 0;
        for (int t = 0; t < T_IN; ++t) {
            s += dur[b * T_IN + t];
            cum[b * T_IN + t] = s;
        }
    }
}

// ---------------------------------------------------------------------------
// Length regulator: dual-store f32 (for output math) + f16 (for conv input).
// ---------------------------------------------------------------------------
__global__ __launch_bounds__(64)
void lenreg_kernel(const float* __restrict__ x, const int* __restrict__ cum,
                   float* __restrict__ xe, half_t* __restrict__ xe16)
{
    const int b = blockIdx.y, frame = blockIdx.x;
    __shared__ int   sidx;
    __shared__ float smask;
    if (threadIdx.x == 0) {
        const int* c = cum + b * T_IN;
        int lo = 0, hi = T_IN;                 // count of c[i] <= frame (side="right")
        while (lo < hi) {
            int mid = (lo + hi) >> 1;
            if (c[mid] <= frame) lo = mid + 1; else hi = mid;
        }
        sidx  = lo > T_IN - 1 ? T_IN - 1 : lo;
        smask = (frame < c[T_IN - 1]) ? 1.0f : 0.0f;
    }
    __syncthreads();
    const float* src = x + ((size_t)b * T_IN + sidx) * D_SZ;
    const size_t ro  = ((size_t)b * MEL + frame) * D_SZ;
    const float  msk = smask;
    for (int i = threadIdx.x; i < D_SZ; i += 64) {
        float v = src[i] * msk;
        xe[ro + i]   = v;
        xe16[ro + i] = (half_t)v;
    }
}

// ---------------------------------------------------------------------------
// xout[b,f,:] = xin[b,f,:] + emb[searchsorted(bins, target, left)]
// optional f16 dual-store for the downstream conv.
// ---------------------------------------------------------------------------
__global__ __launch_bounds__(64)
void addemb_kernel(const float* __restrict__ xin, const float* __restrict__ bins,
                   const float* __restrict__ target, const float* __restrict__ emb,
                   float* __restrict__ xout, half_t* __restrict__ xout16)
{
    const int b = blockIdx.y, frame = blockIdx.x;
    __shared__ int sidx;
    if (threadIdx.x == 0) {
        float t = target[b * MEL + frame];
        int lo = 0, hi = NBINS - 1;            // 255 bin edges; side="left"
        while (lo < hi) {
            int mid = (lo + hi) >> 1;
            if (bins[mid] < t) lo = mid + 1; else hi = mid;
        }
        sidx = lo;                              // 0..255
    }
    __syncthreads();
    const float* e  = emb + (size_t)sidx * D_SZ;
    const size_t ro = ((size_t)b * MEL + frame) * D_SZ;
    for (int i = threadIdx.x; i < D_SZ; i += 64) {
        float v = xin[ro + i] + e[i];
        xout[ro + i] = v;
        if (xout16) xout16[ro + i] = (half_t)v;
    }
}

// ---------------------------------------------------------------------------
extern "C" void kernel_launch(void* const* d_in, const int* in_sizes, int n_in,
                              void* d_out, int out_size, void* d_ws, size_t ws_size,
                              hipStream_t stream)
{
    (void)in_sizes; (void)n_in; (void)out_size; (void)ws_size;

    const float* x  = (const float*)d_in[0];
    const int*   lt = (const int*)d_in[1];
    const float* et = (const float*)d_in[2];
    const float* pt = (const float*)d_in[3];
    // d_in[4] = mel_max_length scalar (2048, compiled in)

#define P(pred, off) ((const float*)d_in[5 + (pred)*10 + (off)])
    const float* ebins = (const float*)d_in[35];
    const float* pbins = (const float*)d_in[36];
    const float* eemb  = (const float*)d_in[37];
    const float* pemb  = (const float*)d_in[38];

    const size_t XE_ELEMS = (size_t)B_SZ * MEL * D_SZ;     // 16,777,216
    const size_t X_ELEMS  = (size_t)B_SZ * T_IN * D_SZ;    //  2,097,152

    char* wsb = (char*)d_ws;
    float*  xe   = (float*)wsb;                                     // 64 MB
    half_t* xe16 = (half_t*)(wsb + XE_ELEMS * 4);                   // 32 MB
    half_t* hA   = (half_t*)(wsb + XE_ELEMS * 4 + XE_ELEMS * 2);    // 32 MB
    half_t* hB   = (half_t*)(wsb + XE_ELEMS * 4 + XE_ELEMS * 4);    // 32 MB
    half_t* x16  = (half_t*)(wsb + XE_ELEMS * 4 + XE_ELEMS * 6);    //  4 MB
    int*    cum  = (int*)(wsb + XE_ELEMS * 4 + XE_ELEMS * 6 + X_ELEMS * 2);

    float* out_xe  = (float*)d_out;                         // [B,MEL,D]
    float* out_dur = out_xe + XE_ELEMS;                     // [B,T_IN]
    float* out_en  = out_dur + (size_t)B_SZ * T_IN;         // [B,MEL]
    float* out_pi  = out_en + (size_t)B_SZ * MEL;           // [B,MEL]

    // ---- duration predictor (on x, T=256) ------------------------------
    f32_to_f16_kernel<<<(int)(X_ELEMS / 256), 256, 0, stream>>>(x, x16, (int)X_ELEMS);
    conv_ln_relu_kernel<<<dim3(T_IN / 16, B_SZ), 128, 0, stream>>>(
        x16, P(0,0), P(0,1), P(0,2), P(0,3), hA, T_IN);
    conv_ln_relu_kernel<<<dim3(T_IN / 16, B_SZ), 128, 0, stream>>>(
        hA, P(0,4), P(0,5), P(0,6), P(0,7), hB, T_IN);
    rowdot_kernel<<<(B_SZ * T_IN) / 8, 256, 0, stream>>>(
        hB, P(0,8), P(0,9), out_dur, B_SZ * T_IN, 1);

    // ---- length regulation --------------------------------------------
    cumsum_kernel<<<B_SZ, 32, 0, stream>>>(lt, cum);
    lenreg_kernel<<<dim3(MEL, B_SZ), 64, 0, stream>>>(x, cum, xe, xe16);

    // ---- energy predictor (on xe, T=2048) ------------------------------
    conv_ln_relu_kernel<<<dim3(MEL / 16, B_SZ), 128, 0, stream>>>(
        xe16, P(1,0), P(1,1), P(1,2), P(1,3), hA, MEL);
    conv_ln_relu_kernel<<<dim3(MEL / 16, B_SZ), 128, 0, stream>>>(
        hA, P(1,4), P(1,5), P(1,6), P(1,7), hB, MEL);
    rowdot_kernel<<<(B_SZ * MEL) / 8, 256, 0, stream>>>(
        hB, P(1,8), P(1,9), out_en, B_SZ * MEL, 0);
    addemb_kernel<<<dim3(MEL, B_SZ), 64, 0, stream>>>(xe, ebins, et, eemb, xe, xe16);

    // ---- pitch predictor (on xe + energy_emb) --------------------------
    conv_ln_relu_kernel<<<dim3(MEL / 16, B_SZ), 128, 0, stream>>>(
        xe16, P(2,0), P(2,1), P(2,2), P(2,3), hA, MEL);
    conv_ln_relu_kernel<<<dim3(MEL / 16, B_SZ), 128, 0, stream>>>(
        hA, P(2,4), P(2,5), P(2,6), P(2,7), hB, MEL);
    rowdot_kernel<<<(B_SZ * MEL) / 8, 256, 0, stream>>>(
        hB, P(2,8), P(2,9), out_pi, B_SZ * MEL, 0);
    addemb_kernel<<<dim3(MEL, B_SZ), 64, 0, stream>>>(
        xe, pbins, pt, pemb, out_xe, (half_t*)nullptr);
#undef P
}
